// Tensor4D_49503793053730
// MI455X (gfx1250) — matrix operand order repaired and checked by reading, not compiled, not run on hardware
//
#include <hip/hip_runtime.h>

typedef __attribute__((ext_vector_type(16))) _Float16 v16h;
typedef __attribute__((ext_vector_type(8)))  _Float16 v8h;
typedef __attribute__((ext_vector_type(8)))  float    v8f;

#define NPTS 524288
#define RES 512
#define PLANE_SZ (RES * RES)                 // elems per channel image
#define PLANE_STRIDE (16 * PLANE_SZ)         // elems per 16-channel plane
#define FRAG_H 512                           // halfs per 32x16 B-fragment
#define NFRAG1 (3 * 16 * 2)                  // planes x n_tiles x k_chunks
#define NFRAG2 (3 * 8)                       // planes x k_chunks
#define FROW 88                              // feature row stride (halfs), 16B aligned, low-conflict
#define SROW 40                              // staging row stride (halfs), 16B aligned, low-conflict
#define WLDS (16 * FROW)                     // per-wave LDS halfs (staging overlaps features)

__device__ __forceinline__ v16h cat8(v8h lo, v8h hi) {
  v16h r;
#pragma unroll
  for (int i = 0; i < 8; ++i) { r[i] = lo[i]; r[i + 8] = hi[i]; }
  return r;
}

// Pre-swizzle cw1 (3,256,48) and cw2 (3,16,256) into WMMA B-fragment layout:
// B element e of lane l maps to K = (e&7) + 16*(e>>3) + 8*(l>=16) (+32*kchunk), N = l%16.
__global__ __launch_bounds__(256) void prep_weights(const float* __restrict__ cw1,
                                                    const float* __restrict__ cw2,
                                                    _Float16* __restrict__ ws1,
                                                    _Float16* __restrict__ ws2) {
  int tid = blockIdx.x * 256 + threadIdx.x;
  const int n1 = NFRAG1 * FRAG_H;            // 49152
  if (tid < n1) {
    int frag = tid >> 9;
    int r = tid & 511;
    int lane = r >> 4, e = r & 15;
    int kc = frag & 1;
    int nt = (frag >> 1) & 15;
    int p  = frag >> 5;
    int k = (e & 7) + 16 * (e >> 3) + 8 * (lane >> 4) + 32 * kc;
    int n = nt * 16 + (lane & 15);
    float v = (k < 48) ? cw1[(p * 256 + n) * 48 + k] : 0.0f;   // pad K 48 -> 64
    ws1[tid] = (_Float16)v;
  } else {
    int u = tid - n1;
    if (u >= NFRAG2 * FRAG_H) return;
    int frag = u >> 9;
    int r = u & 511;
    int lane = r >> 4, e = r & 15;
    int kc = frag & 7;
    int p  = frag >> 3;
    int k = (e & 7) + 16 * (e >> 3) + 8 * (lane >> 4) + 32 * kc;
    int n = lane & 15;
    ws2[u] = (_Float16)cw2[(p * 16 + n) * 256 + k];
  }
}

__device__ __forceinline__ void bilin(const float* __restrict__ img, float cx, float cy,
                                      int ch0, int nch, _Float16* __restrict__ dst) {
  float ix = (cx + 1.0f) * 0.5f * (float)(RES - 1);
  float iy = (cy + 1.0f) * 0.5f * (float)(RES - 1);
  float fx = floorf(ix), fy = floorf(iy);
  float wE = ix - fx, wW = fx + 1.0f - ix;
  float wS = iy - fy, wN = fy + 1.0f - iy;
  float nw = wW * wN, ne = wE * wN, sw = wW * wS, se = wE * wS;
  int x0 = (int)fminf(fmaxf(fx, 0.0f), (float)(RES - 1));
  int y0 = (int)fminf(fmaxf(fy, 0.0f), (float)(RES - 1));
  int x1 = (int)fminf(fmaxf(fx + 1.0f, 0.0f), (float)(RES - 1));
  int y1 = (int)fminf(fmaxf(fy + 1.0f, 0.0f), (float)(RES - 1));
  int i00 = y0 * RES + x0, i01 = y0 * RES + x1;
  int i10 = y1 * RES + x0, i11 = y1 * RES + x1;
  const float* im = img + (size_t)ch0 * PLANE_SZ;
#pragma unroll 4
  for (int c = 0; c < nch; ++c, im += PLANE_SZ) {
    float v = nw * im[i00] + ne * im[i01] + sw * im[i10] + se * im[i11];
    dst[c] = (_Float16)v;
  }
}

// One wave32 = 16 points; block = 8 waves = 128 points.
// Per plane: gather 16x48(->64) f16 features to LDS, 2 WMMA k-chunks x 16 N-tiles
// for layer 1 (H 16x256), each N-tile pair staged via LDS (C-layout -> A-layout)
// and consumed by one layer-2 WMMA against W2 fragments. 120 WMMAs per wave.
__global__ __launch_bounds__(256) void fused_planes(const float* __restrict__ xyz,
                                                    const float* __restrict__ temb,
                                                    const float* __restrict__ space_planes,
                                                    const float* __restrict__ tspace_planes,
                                                    const float* __restrict__ cb1,
                                                    const float* __restrict__ cb2,
                                                    const _Float16* __restrict__ wf1,
                                                    const _Float16* __restrict__ wf2,
                                                    float* __restrict__ out) {
  __shared__ _Float16 lds[8 * WLDS];
  const int lane = threadIdx.x & 31;
  const int wv   = threadIdx.x >> 5;
  const int pt   = lane & 15;
  const int half = lane >> 4;
  const int gbase = blockIdx.x * 128 + wv * 16;
  const int gpt   = gbase + pt;
  _Float16* myLds = lds + wv * WLDS;
  _Float16* frow  = myLds + pt * FROW;

  const float X = xyz[3 * gpt + 0];
  const float Y = xyz[3 * gpt + 1];
  const float Z = xyz[3 * gpt + 2];
  const float T = temb[0];

  for (int p = 0; p < 3; ++p) {
    // matMode: p0 -> (y,z), p1 -> (x,z), p2 -> (x,y)
    const float s0 = (p == 0) ? Y : X;
    const float s1 = (p == 2) ? Y : Z;
    const float* sp  = space_planes  + (size_t)p * PLANE_STRIDE;
    const float* tp0 = tspace_planes + (size_t)(2 * p) * PLANE_STRIDE;
    const float* tp1 = tspace_planes + (size_t)(2 * p + 1) * PLANE_STRIDE;

    // Feature gather: feats 0-15 = space(s0,s1), 16-31 = t0(T,s0), 32-47 = t1(T,s1), 48-63 = 0
    if (half == 0) {
      bilin(sp,  s0, s1, 0, 16, frow + 0);
      bilin(tp0, T,  s0, 0, 8,  frow + 16);
#pragma unroll
      for (int j = 0; j < 8; ++j) frow[48 + j] = (_Float16)0.0f;
    } else {
      bilin(tp0, T, s0, 8, 8,  frow + 24);
      bilin(tp1, T, s1, 0, 16, frow + 32);
#pragma unroll
      for (int j = 0; j < 8; ++j) frow[56 + j] = (_Float16)0.0f;
    }

    // Layer-1 A fragments (A layout: row M = lane%16, k = (e&7) + 16*(e>>3) + 8*half + 32*kc)
    v16h a0 = cat8(*(const v8h*)(frow + 8 * half),
                   *(const v8h*)(frow + 16 + 8 * half));
    v16h a1 = cat8(*(const v8h*)(frow + 32 + 8 * half),
                   *(const v8h*)(frow + 48 + 8 * half));

    v8f acc2 = {0.f, 0.f, 0.f, 0.f, 0.f, 0.f, 0.f, 0.f};
#pragma unroll
    for (int t = 0; t < 8; ++t) {
#pragma unroll
      for (int s = 0; s < 2; ++s) {
        const int nt = 2 * t + s;
        const _Float16* fb = wf1 + (size_t)((p * 16 + nt) * 2) * FRAG_H + lane * 16;
        v16h b0 = *(const v16h*)(fb);
        v16h b1 = *(const v16h*)(fb + FRAG_H);
        v8f acc = {0.f, 0.f, 0.f, 0.f, 0.f, 0.f, 0.f, 0.f};
        acc = __builtin_amdgcn_wmma_f32_16x16x32_f16(false, a0, false, b0, (short)0, acc, false, false);
        acc = __builtin_amdgcn_wmma_f32_16x16x32_f16(false, a1, false, b1, (short)0, acc, false, false);
        const float bias = cb1[p * 256 + nt * 16 + pt];
#pragma unroll
        for (int r = 0; r < 8; ++r) {
          float v = acc[r] + bias;
          v = (v >= 0.0f) ? v : 0.01f * v;
          // stage H tile: row = r + 8*half, local k = s*16 + col(pt)
          myLds[(r + 8 * half) * SROW + s * 16 + pt] = (_Float16)v;
        }
      }
      // consume staged 16x32 H block as layer-2 A fragment (k-chunk t)
      v16h a2 = cat8(*(const v8h*)(myLds + pt * SROW + 8 * half),
                     *(const v8h*)(myLds + pt * SROW + 16 + 8 * half));
      v16h b2 = *(const v16h*)(wf2 + (size_t)(p * 8 + t) * FRAG_H + lane * 16);
      acc2 = __builtin_amdgcn_wmma_f32_16x16x32_f16(false, a2, false, b2, (short)0, acc2, false, false);
    }

    const float bias2 = cb2[p * 16 + pt];
#pragma unroll
    for (int r = 0; r < 8; ++r) {
      float v = acc2[r] + bias2;
      v = (v >= 0.0f) ? v : 0.01f * v;
      out[(size_t)(gbase + r + 8 * half) * 48 + p * 16 + pt] = v;
    }
  }
}

extern "C" void kernel_launch(void* const* d_in, const int* in_sizes, int n_in,
                              void* d_out, int out_size, void* d_ws, size_t ws_size,
                              hipStream_t stream) {
  (void)in_sizes; (void)n_in; (void)out_size; (void)ws_size;
  const float* xyz  = (const float*)d_in[0];   // (NP,3)
  const float* temb = (const float*)d_in[1];   // scalar
  const float* sp   = (const float*)d_in[2];   // (3,16,512,512)
  const float* tsp  = (const float*)d_in[3];   // (6,16,512,512)
  const float* cw1  = (const float*)d_in[4];   // (3,256,48)
  const float* cb1  = (const float*)d_in[5];   // (3,256)
  const float* cw2  = (const float*)d_in[6];   // (3,16,256)
  const float* cb2  = (const float*)d_in[7];   // (3,16)
  // d_in[8] = fid (unused)

  _Float16* ws1 = (_Float16*)d_ws;                 // 49152 halfs
  _Float16* ws2 = ws1 + (size_t)NFRAG1 * FRAG_H;   // 12288 halfs

  const int prep_total = (NFRAG1 + NFRAG2) * FRAG_H;      // 61440
  prep_weights<<<(prep_total + 255) / 256, 256, 0, stream>>>(cw1, cw2, ws1, ws2);

  fused_planes<<<NPTS / 128, 256, 0, stream>>>(xyz, temb, sp, tsp, cb1, cb2,
                                               ws1, ws2, (float*)d_out);
}
